// FPNDAM_51951924412992
// MI455X (gfx1250) — matrix operand order repaired and checked by reading, not compiled, not run on hardware
//
#include <hip/hip_runtime.h>
#include <math.h>

// ---------------------------------------------------------------------------
// CDNA5 (gfx1250) FPN-DAM implementation, v3.
// GEMM-shaped stages (1x1 lateral convs, 3x3 output convs) run on
// v_wmma_f32_16x16x32_bf16 (fp32 accumulate). Weights are pre-converted once
// to bf16 in a fragment-friendly layout and streamed straight from L2 as
// A-fragments; only activations go through LDS (transposed + packed bf16).
// v3: fragment loads batched into distinct registers ahead of the WMMA chain
// so each K-chunk pays a single s_wait_dscnt instead of one per WMMA.
// ---------------------------------------------------------------------------

typedef __attribute__((ext_vector_type(16))) __bf16 v16bf;
typedef __attribute__((ext_vector_type(8)))  float  v8f;

#define EPS_ 1e-7f
#define NB_ 4          // batch
#define NC_ 256        // FPN channel count

__device__ __forceinline__ v8f zero8() {
    v8f z = {0.f,0.f,0.f,0.f,0.f,0.f,0.f,0.f};
    return z;
}

// A-matrix fragment (16x32 bf16, M x K). rowp points at a K-major row of 32
// bf16 (LDS or global). ISA layout: lanes 0-15 hold K{0-7,16-23}, lanes 16-31
// hold K{8-15,24-31}: slot s<8 -> K=8*laneHi+s, slot s>=8 -> K=16+8*laneHi+s-8.
// Two contiguous 16 B reads per lane.
__device__ __forceinline__ v16bf load_frag_A(const __bf16* rowp, int laneHi) {
    v16bf f;
    const __bf16* p0 = rowp + 8 * laneHi;
    const __bf16* p1 = rowp + 16 + 8 * laneHi;
#pragma unroll
    for (int s = 0; s < 8; ++s) { f[s] = p0[s]; f[s + 8] = p1[s]; }
    return f;
}

// B-matrix fragment (32x16 bf16, K x N). rowp = K-major row of 32 bf16 for
// column n = lane&15. slot s<8 -> K = 8*laneHi + s, s>=8 -> K = 16+8*laneHi+s-8.
__device__ __forceinline__ v16bf load_frag_B(const __bf16* rowp, int laneHi) {
    v16bf f;
    const __bf16* p0 = rowp + 8 * laneHi;
    const __bf16* p1 = rowp + 16 + 8 * laneHi;
#pragma unroll
    for (int s = 0; s < 8; ++s) { f[s] = p0[s]; f[s + 8] = p1[s]; }
    return f;
}

// ---------------------------------------------------------------------------
// One-time weight conversions (bf16, fragment-friendly layouts in d_ws).
// ---------------------------------------------------------------------------
__global__ __launch_bounds__(256)
void k_cvt(const float* __restrict__ src, __bf16* __restrict__ dst, int n)
{
    const int i = blockIdx.x * 256 + threadIdx.x;
    if (i < n) dst[i] = (__bf16)src[i];
}

// fw [co][ci][3][3] (f32) -> dst [tap][co][ci] (bf16); 9*256*256 elems.
// e decomposes exactly as the destination index.
__global__ __launch_bounds__(256)
void k_cvt_conv(const float* __restrict__ fw, __bf16* __restrict__ dst)
{
    const int e   = blockIdx.x * 256 + threadIdx.x;
    const int ci  = e & 255;
    const int co  = (e >> 8) & 255;
    const int tap = e >> 16;
    dst[e] = (__bf16)fw[(size_t)co * (NC_ * 9) + (size_t)ci * 9 + tap];
}

// ---------------------------------------------------------------------------
// Lateral 1x1 conv as GEMM:  out[b,co,p] = sum_ci W[co,ci] * X[b,ci,p] + bias
// Block tile: 64 co x 256 px, K chunks of 32. 8 waves: wave = (mSub<<1)|nHalf,
// 8 WMMA per wave per K chunk. A-fragments straight from global bf16 weights
// (L2-resident); activations transposed to [px][ci] bf16 in LDS via paired
// float4 row loads (coalesced b128) + packed bf16 stores.
// ---------------------------------------------------------------------------
__global__ __launch_bounds__(256)
void k_lateral(const float* __restrict__ x, const __bf16* __restrict__ wbf,
               const float* __restrict__ bias, float* __restrict__ out,
               int Cin, int HW)
{
    __shared__ __bf16 Blds[256][32];   // [px][ci]  16 KB

    const int t      = threadIdx.x;
    const int wid    = t >> 5;
    const int lane   = t & 31;
    const int lane15 = lane & 15;
    const int laneHi = lane >> 4;

    const int coBase = blockIdx.y * 64;
    const int pBase  = blockIdx.x * 256;
    const int b      = blockIdx.z;

    const float* xb = x + (size_t)b * Cin * HW;

    const int mSub  = wid >> 1;         // co subtile 0..3
    const int nHalf = (wid & 1) * 128;  // pixel half

    v8f acc[8];
#pragma unroll
    for (int s = 0; s < 8; ++s) acc[s] = zero8();

    // this wave-lane's A row (K-major, bf16, global)
    const __bf16* arow = wbf + (size_t)(coBase + mSub * 16 + lane15) * Cin;

    for (int kc = 0; kc < Cin; kc += 32) {
        // ---- stage B: 32 ci x 256 px. 1024 (ci-pair, px4) units, 4/thread.
        // consecutive threads -> consecutive float4 within a row (coalesced).
#pragma unroll
        for (int j = 0; j < 4; ++j) {
            const int e  = t + 256 * j;
            const int k2 = e >> 6;      // ci pair 0..15
            const int p4 = e & 63;      // px group 0..63
            const float* r0 = xb + (size_t)(kc + 2 * k2) * HW + pBase + 4 * p4;
            const float4 ra = *reinterpret_cast<const float4*>(r0);
            const float4 rb = *reinterpret_cast<const float4*>(r0 + HW);
            __bf16* d0 = &Blds[4 * p4 + 0][2 * k2];
            d0[0]  = (__bf16)ra.x; d0[1]  = (__bf16)rb.x;
            d0[32] = (__bf16)ra.y; d0[33] = (__bf16)rb.y;
            d0[64] = (__bf16)ra.z; d0[65] = (__bf16)rb.z;
            d0[96] = (__bf16)ra.w; d0[97] = (__bf16)rb.w;
        }
        // prefetch next K chunk of activations (global_prefetch_b8)
        if (kc + 32 < Cin)
            __builtin_prefetch(xb + (size_t)(kc + 32 + (t >> 4)) * HW + pBase + ((t & 15) << 4), 0, 1);
        __syncthreads();

        // batch all fragment loads, then a single WMMA chain
        const v16bf a = load_frag_A(arow + kc, laneHi);
        v16bf bfr[8];
#pragma unroll
        for (int s = 0; s < 8; ++s)
            bfr[s] = load_frag_B(&Blds[nHalf + s * 16 + lane15][0], laneHi);
#pragma unroll
        for (int s = 0; s < 8; ++s)
            acc[s] = __builtin_amdgcn_wmma_f32_16x16x32_bf16(
                false, a, false, bfr[s], (short)0, acc[s], false, false);
        __syncthreads();
    }

    // C/D layout: row = v + 8*laneHi, col = lane&15
#pragma unroll
    for (int s = 0; s < 8; ++s) {
#pragma unroll
        for (int v = 0; v < 8; ++v) {
            const int co = coBase + mSub * 16 + v + 8 * laneHi;
            const int p  = pBase + nHalf + s * 16 + lane15;
            out[((size_t)b * NC_ + co) * HW + p] = acc[s][v] + bias[co];
        }
    }
}

// ---------------------------------------------------------------------------
// 3x3 conv (256ci -> 64co per block) as 9 tap-GEMM accumulations, pad=1.
// Weights come straight from global bf16 [tap][co][ci] (L2-hot A-fragments);
// only the input patch is staged in LDS (~6.5 KB). (TCOLS/16)*TROWS == 2 so
// 8 waves cover 4(m) x 2(n) subtiles, one accumulator per wave.
// ---------------------------------------------------------------------------
template<int TCOLS, int TROWS>
__global__ __launch_bounds__(256)
void k_conv3x3(const float* __restrict__ x, const __bf16* __restrict__ wbf,
               const float* __restrict__ bias, float* __restrict__ out,
               int H, int W)
{
    static_assert((TCOLS / 16) * TROWS == 2, "need 8 subtiles for 8 waves");
    constexpr int NSUBC = TCOLS / 16;
    constexpr int PROWS = TROWS + 2;
    constexpr int PCOLS = TCOLS + 2;

    __shared__ __bf16 Xp[PROWS][PCOLS][32];   // input patch, ci innermost

    const int t      = threadIdx.x;
    const int wid    = t >> 5;
    const int lane   = t & 31;
    const int lane15 = lane & 15;
    const int laneHi = lane >> 4;

    const int tilesPerRow = W / TCOLS;
    const int rowTiles    = H / TROWS;
    const int bx = blockIdx.x;
    const int tx = bx % tilesPerRow;
    const int r0 = ((bx / tilesPerRow) % rowTiles) * TROWS;
    const int b  = bx / (tilesPerRow * rowTiles);
    const int c0 = tx * TCOLS;
    const int coBase = blockIdx.y * 64;

    const int mSub   = wid >> 1;
    const int nSub   = wid & 1;
    const int rowOff = nSub / NSUBC;
    const int colOff = (nSub % NSUBC) * 16;

    const size_t planeHW = (size_t)H * W;
    const float* xb = x + (size_t)b * NC_ * planeHW;

    // this wave-lane's A row base (add tap*NC_*NC_ + kc per fragment)
    const __bf16* arowBase = wbf + (size_t)(coBase + mSub * 16 + lane15) * NC_;

    v8f acc = zero8();

    for (int kc = 0; kc < NC_; kc += 32) {
        // ---- stage input patch, ci pairs packed (zero-padded halo) ----
        constexpr int NPAIR = PROWS * PCOLS * 16;
        for (int e = t; e < NPAIR; e += 256) {
            const int ci2  = e / (PROWS * PCOLS);   // 0..15
            const int rem  = e % (PROWS * PCOLS);
            const int dyp  = rem / PCOLS;
            const int colp = rem % PCOLS;
            const int row  = r0 + dyp - 1;
            const int col  = c0 + colp - 1;
            float va = 0.f, vb = 0.f;
            if ((unsigned)row < (unsigned)H && (unsigned)col < (unsigned)W) {
                const float* p = xb + (size_t)(kc + 2 * ci2) * planeHW
                                    + (size_t)row * W + col;
                va = p[0];
                vb = p[planeHW];
            }
            Xp[dyp][colp][2 * ci2]     = (__bf16)va;
            Xp[dyp][colp][2 * ci2 + 1] = (__bf16)vb;
        }
        __syncthreads();

        // batch all 9 A- and B-fragment loads, then 9 chained WMMAs:
        // global b128 clause + ds_load clause -> single waits -> wmma run
        v16bf afr[9], bfr[9];
#pragma unroll
        for (int tap = 0; tap < 9; ++tap)
            afr[tap] = load_frag_A(arowBase + (size_t)tap * NC_ * NC_ + kc, laneHi);
#pragma unroll
        for (int tap = 0; tap < 9; ++tap)
            bfr[tap] = load_frag_B(&Xp[tap / 3 + rowOff][colOff + lane15 + tap % 3][0], laneHi);
#pragma unroll
        for (int tap = 0; tap < 9; ++tap)
            acc = __builtin_amdgcn_wmma_f32_16x16x32_bf16(
                false, afr[tap], false, bfr[tap], (short)0, acc, false, false);
        __syncthreads();
    }

#pragma unroll
    for (int v = 0; v < 8; ++v) {
        const int co  = coBase + mSub * 16 + v + 8 * laneHi;
        const int row = r0 + rowOff;
        const int col = c0 + colOff + lane15;
        out[((size_t)b * NC_ + co) * planeHW + (size_t)row * W + col] = acc[v] + bias[co];
    }
}

// ---------------------------------------------------------------------------
// basic = relu(conv3x3(lat, rw, rb, pad=1))[:,0]   (1 output channel)
// ---------------------------------------------------------------------------
__global__ __launch_bounds__(256)
void k_basic(const float* __restrict__ lat, const float* __restrict__ rw,
             const float* __restrict__ rb, float* __restrict__ basic, int n)
{
    const int idx = blockIdx.x * 256 + threadIdx.x;
    if (idx >= NB_ * n * n) return;
    const int x = idx % n;
    const int y = (idx / n) % n;
    const int b = idx / (n * n);
    const float* latb = lat + (size_t)b * NC_ * n * n;
    float s = rb[0];
    for (int ci = 0; ci < NC_; ++ci) {
        const float* p  = latb + (size_t)ci * n * n;
        const float* wp = rw + ci * 9;
#pragma unroll
        for (int dy = 0; dy < 3; ++dy) {
            const int yy = y + dy - 1;
            if ((unsigned)yy >= (unsigned)n) continue;
#pragma unroll
            for (int dx = 0; dx < 3; ++dx) {
                const int xx = x + dx - 1;
                if ((unsigned)xx >= (unsigned)n) continue;
                s += wp[dy * 3 + dx] * p[yy * n + xx];
            }
        }
    }
    basic[idx] = fmaxf(s, 0.f);
}

// ---------------------------------------------------------------------------
// Per-image stats: row/col means of channel 0; min/max of the rank-1 map
// row*col (exact from the 4 endpoint products); min/max of basic.
// ---------------------------------------------------------------------------
__global__ __launch_bounds__(256)
void k_stats(const float* __restrict__ lat, const float* __restrict__ basic,
             float* __restrict__ rowm, float* __restrict__ colm,
             float* __restrict__ scal, int n)
{
    __shared__ float sRow[128], sCol[128];
    __shared__ float sMn[256], sMx[256];
    const int b = blockIdx.x;
    const int t = threadIdx.x;
    const float* c0p = lat + (size_t)b * NC_ * n * n;  // channel 0 plane

    if (t < n) {
        float s = 0.f;
        for (int xx = 0; xx < n; ++xx) s += c0p[t * n + xx];
        const float m = s / n;
        sRow[t] = m; rowm[b * 128 + t] = m;
        float s2 = 0.f;
        for (int yy = 0; yy < n; ++yy) s2 += c0p[yy * n + t];
        const float m2 = s2 / n;
        sCol[t] = m2; colm[b * 128 + t] = m2;
    }
    const float* bs = basic + (size_t)b * n * n;
    float mn = 3.4e38f, mx = -3.4e38f;
    for (int i = t; i < n * n; i += 256) {
        const float v = bs[i];
        mn = fminf(mn, v); mx = fmaxf(mx, v);
    }
    sMn[t] = mn; sMx[t] = mx;
    __syncthreads();
    for (int off = 128; off > 0; off >>= 1) {
        if (t < off) {
            sMn[t] = fminf(sMn[t], sMn[t + off]);
            sMx[t] = fmaxf(sMx[t], sMx[t + off]);
        }
        __syncthreads();
    }
    if (t == 0) {
        float rmn = sRow[0], rmx = sRow[0], cmn = sCol[0], cmx = sCol[0];
        for (int i = 1; i < n; ++i) {
            rmn = fminf(rmn, sRow[i]); rmx = fmaxf(rmx, sRow[i]);
            cmn = fminf(cmn, sCol[i]); cmx = fmaxf(cmx, sCol[i]);
        }
        const float p1 = rmn * cmn, p2 = rmn * cmx, p3 = rmx * cmn, p4 = rmx * cmx;
        const float amn = fminf(fminf(p1, p2), fminf(p3, p4));
        const float amx = fmaxf(fmaxf(p1, p2), fmaxf(p3, p4));
        scal[b * 4 + 0] = amn;
        scal[b * 4 + 1] = amx - amn;
        scal[b * 4 + 2] = sMn[0];
        scal[b * 4 + 3] = sMx[0] - sMn[0];
    }
}

// ---------------------------------------------------------------------------
// lat += cos((avg_n - basic_n)*pi/2) * upsample2x(latUp)
// ---------------------------------------------------------------------------
__global__ __launch_bounds__(256)
void k_fuse(float* __restrict__ lat, const float* __restrict__ up,
            const float* __restrict__ basic, const float* __restrict__ rowm,
            const float* __restrict__ colm, const float* __restrict__ scal, int n)
{
    const size_t idx = (size_t)blockIdx.x * 256 + threadIdx.x;
    const size_t total = (size_t)NB_ * NC_ * n * n;
    if (idx >= total) return;
    const int x = (int)(idx % n);
    const int y = (int)((idx / n) % n);
    const int c = (int)((idx / ((size_t)n * n)) % NC_);
    const int b = (int)(idx / ((size_t)n * n * NC_));

    const float amn = scal[b * 4 + 0], arng = scal[b * 4 + 1];
    const float bmn = scal[b * 4 + 2], brng = scal[b * 4 + 3];
    const float av = (rowm[b * 128 + y] * colm[b * 128 + x] - amn) / (arng + EPS_);
    const float bv = (basic[(size_t)b * n * n + (size_t)y * n + x] - bmn) / (brng + EPS_);
    const float dist = cosf((av - bv) * 1.5707963267948966f);

    const int hn = n >> 1;
    const float u = up[(((size_t)b * NC_ + c) * hn + (y >> 1)) * hn + (x >> 1)];
    lat[idx] += dist * u;
}

// out4 = out3[:, :, ::2, ::2]   (out3 is 16x16 -> 8x8)
__global__ __launch_bounds__(256)
void k_sub2(const float* __restrict__ src, float* __restrict__ dst)
{
    const int idx = blockIdx.x * 256 + threadIdx.x;  // 4*256*8*8 = 65536
    const int x  = idx & 7;
    const int y  = (idx >> 3) & 7;
    const int bc = idx >> 6;
    dst[idx] = src[(size_t)bc * 256 + (size_t)(y * 2) * 16 + x * 2];
}

// ---------------------------------------------------------------------------
extern "C" void kernel_launch(void* const* d_in, const int* in_sizes, int n_in,
                              void* d_out, int out_size, void* d_ws, size_t ws_size,
                              hipStream_t stream)
{
    (void)in_sizes; (void)n_in; (void)out_size; (void)ws_size;

    // setup_inputs() dict order:
    // [c2,lw0,lb0,fw0,fb0, c3,lw1,lb1,fw1,fb1, c4,lw2,lb2,fw2,fb2,
    //  c5,lw3,lb3,fw3,fb3, rw0,rb0,rw1,rb1,rw2,rb2]
    const float* cIn[4] = { (const float*)d_in[0],  (const float*)d_in[5],
                            (const float*)d_in[10], (const float*)d_in[15] };
    const float* lw[4]  = { (const float*)d_in[1],  (const float*)d_in[6],
                            (const float*)d_in[11], (const float*)d_in[16] };
    const float* lb[4]  = { (const float*)d_in[2],  (const float*)d_in[7],
                            (const float*)d_in[12], (const float*)d_in[17] };
    const float* fw[4]  = { (const float*)d_in[3],  (const float*)d_in[8],
                            (const float*)d_in[13], (const float*)d_in[18] };
    const float* fb[4]  = { (const float*)d_in[4],  (const float*)d_in[9],
                            (const float*)d_in[14], (const float*)d_in[19] };
    const float* rw[3]  = { (const float*)d_in[20], (const float*)d_in[22],
                            (const float*)d_in[24] };
    const float* rb[3]  = { (const float*)d_in[21], (const float*)d_in[23],
                            (const float*)d_in[25] };

    const int HWs[4]  = { 128, 64, 32, 16 };
    const int CINs[4] = { 256, 512, 1024, 2048 };

    // ---- workspace layout ----
    float* ws = (float*)d_ws;
    float* lat[4];
    lat[0] = ws;                       // 4*256*128*128 = 16777216 f32
    lat[1] = lat[0] + 16777216;        // 4194304
    lat[2] = lat[1] + 4194304;         // 1048576
    lat[3] = lat[2] + 1048576;         // 262144
    float* basic = lat[3] + 262144;    // up to 4*128*128 = 65536
    float* rowm  = basic + 65536;      // 4*128
    float* colm  = rowm + 512;         // 4*128
    float* scal  = colm + 512;         // 4*4

    // bf16 weight regions (converted once per launch)
    __bf16* wb = (__bf16*)(scal + 16);
    __bf16* lwb[4];
    lwb[0] = wb;                       // 256*256
    lwb[1] = lwb[0] + 65536;           // 256*512
    lwb[2] = lwb[1] + 131072;          // 256*1024
    lwb[3] = lwb[2] + 262144;          // 256*2048
    __bf16* cwb[4];
    cwb[0] = lwb[3] + 524288;          // 9*256*256 each
    cwb[1] = cwb[0] + 589824;
    cwb[2] = cwb[1] + 589824;
    cwb[3] = cwb[2] + 589824;

    float* o0 = (float*)d_out;
    float* o1 = o0 + 16777216;
    float* o2 = o1 + 4194304;
    float* o3 = o2 + 1048576;
    float* o4 = o3 + 262144;

    // 0) weight conversions (bf16, fragment-friendly layouts)
    for (int i = 0; i < 4; ++i) {
        const int nlw = NC_ * CINs[i];
        k_cvt<<<nlw / 256, 256, 0, stream>>>(lw[i], lwb[i], nlw);
        k_cvt_conv<<<589824 / 256, 256, 0, stream>>>(fw[i], cwb[i]);
    }

    // 1) laterals (1x1 conv GEMMs, bf16 WMMA)
    for (int i = 0; i < 4; ++i) {
        const int HW = HWs[i] * HWs[i];
        dim3 grid(HW / 256, NC_ / 64, NB_);
        k_lateral<<<grid, 256, 0, stream>>>(cIn[i], lwb[i], lb[i], lat[i], CINs[i], HW);
    }

    // 2) DAM top-down: lev = 2,1,0 corresponds to reference i = 3,2,1
    for (int lev = 2; lev >= 0; --lev) {
        const int n = HWs[lev];
        k_basic<<<(NB_ * n * n) / 256, 256, 0, stream>>>(lat[lev], rw[lev], rb[lev], basic, n);
        k_stats<<<NB_, 256, 0, stream>>>(lat[lev], basic, rowm, colm, scal, n);
        k_fuse<<<(NB_ * NC_ * n * n) / 256, 256, 0, stream>>>(
            lat[lev], lat[lev + 1], basic, rowm, colm, scal, n);
    }

    // 3) output 3x3 convs (bf16 WMMA, 9-tap GEMM accumulation)
    k_conv3x3<32, 1><<<dim3(NB_ * 128 * (128 / 32), NC_ / 64), 256, 0, stream>>>(
        lat[0], cwb[0], fb[0], o0, 128, 128);
    k_conv3x3<32, 1><<<dim3(NB_ * 64 * (64 / 32), NC_ / 64), 256, 0, stream>>>(
        lat[1], cwb[1], fb[1], o1, 64, 64);
    k_conv3x3<32, 1><<<dim3(NB_ * 32 * (32 / 32), NC_ / 64), 256, 0, stream>>>(
        lat[2], cwb[2], fb[2], o2, 32, 32);
    k_conv3x3<16, 2><<<dim3(NB_ * (16 / 2) * 1, NC_ / 64), 256, 0, stream>>>(
        lat[3], cwb[3], fb[3], o3, 16, 16);

    // 4) stride-2 subsample
    k_sub2<<<65536 / 256, 256, 0, stream>>>(o3, o4);
}
